// MemoryAlgorithm_15882789060892
// MI455X (gfx1250) — matrix operand order repaired and checked by reading, not compiled, not run on hardware
//
#include <hip/hip_runtime.h>
#include <hip/hip_bf16.h>
#include <cstdint>
#include <math.h>

// ---------------------------------------------------------------------------
// MI455X / gfx1250 implementation.
// - All GEMMs run through v_wmma_f32_16x16x32_f16 (f16 in, f32 accumulate).
// - Tile staging uses CDNA5 async global->LDS DMA (ASYNCcnt) with a
//   double-buffered K loop in the GEMM.
// NOTE: the reference's "surprise" renormalization is an exact identity
// (w*(1+s)/sum(w*(1+s)) == w because softmax rows sum to 1), so it is elided.
// ---------------------------------------------------------------------------

typedef __attribute__((ext_vector_type(16))) _Float16 v16h;
typedef __attribute__((ext_vector_type(8)))  _Float16 v8h;
typedef __attribute__((ext_vector_type(8)))  float    v8f;

#define DEV_ __device__ __forceinline__

static constexpr int B_  = 4;
static constexpr int S_  = 2048;
static constexpr int H_  = 1024;
static constexpr int M_  = 512;
static constexpr int NH_ = 16;
static constexpr int HD_ = 64;
static constexpr int HQ_ = 256;
static constexpr int ROWS  = B_ * S_;   // 8192
static constexpr int MROWS = B_ * M_;   // 2048

// ---- CDNA5 async global->LDS copy (ASYNCcnt-tracked DMA) ------------------
// Per-lane: LDS[vdst] = MEM[vaddr], 16 bytes. Generic LDS pointers on
// gfx1250 carry the LDS byte offset in the low 32 bits (aperture layout).
DEV_ void async_b128(void* lds_dst, const void* gsrc) {
  const uint32_t l = (uint32_t)(uintptr_t)lds_dst;
  asm volatile("global_load_async_to_lds_b128 %0, %1, off"
               :: "v"(l), "v"(gsrc) : "memory");
}
DEV_ void wait_async0() {
  asm volatile("s_wait_asynccnt 0x0" ::: "memory");
}

// ---- WMMA fragment helpers (per CDNA5 ISA 7.12.2 16-bit layouts) ----------
// A (16x32, f16): lanes 0-15 hold M=0-15 with K={0..7,16..23}; lanes 16-31
// hold the same rows with K={8..15,24..31}.
DEV_ v16h load_a_frag(const _Float16* base, int stride, int lane) {
  const int row = lane & 15;
  const int ho  = (lane >> 4) << 3;  // 0 or 8 halves
  union { v16h v; v8h h[2]; } u;
  u.h[0] = *(const v8h*)(base + row * stride + ho);
  u.h[1] = *(const v8h*)(base + row * stride + 16 + ho);
  return u.v;
}
// B (32x16, f16): lanes 0-15 hold N=0-15 with K={0..7,16..23}; lanes 16-31
// hold the same columns with K={8..15,24..31}.
DEV_ v16h load_b_frag(const _Float16* base, int stride, int lane) {
  const int col = lane & 15;
  const int ho  = (lane >> 4) << 3;
  v16h v;
#pragma unroll
  for (int i = 0; i < 8; ++i) {
    v[i]     = base[(ho + i) * stride + col];
    v[8 + i] = base[(16 + ho + i) * stride + col];
  }
  return v;
}
DEV_ v8f wmma_f16(v16h a, v16h b, v8f c) {
  return __builtin_amdgcn_wmma_f32_16x16x32_f16(false, a, false, b,
                                                (short)0, c, false, false);
}
DEV_ float gelu_exact(float x) {
  return 0.5f * x * (1.0f + erff(x * 0.70710678118654752f));
}

// ---------------------------------------------------------------------------
// Generic elementwise kernels
// ---------------------------------------------------------------------------
__global__ void cast_kernel(const float* __restrict__ src,
                            _Float16* __restrict__ dst,
                            int n, int cols, int dstride) {
  const int i = blockIdx.x * blockDim.x + threadIdx.x;
  if (i >= n) return;
  const int r = i / cols, c = i - r * cols;
  dst[(size_t)r * dstride + c] = (_Float16)src[i];
}

__global__ void zero_kernel(float* __restrict__ p, int n) {
  const int i = blockIdx.x * blockDim.x + threadIdx.x;
  if (i < n) p[i] = 0.0f;
}

// ---------------------------------------------------------------------------
// Tiled WMMA GEMM: C(MxN) = epi( A(MxK,f16,lda) @ W(KxN,f16,row-major) + bias )
// Block tile 128x128x32, 256 threads = 8 waves in a 4(M)x2(N) grid,
// each wave computes a 32x64 strip = 2x4 16x16 WMMA tiles.
// Double-buffered LDS, staged with async global->LDS DMA: one barrier/K-step.
// ---------------------------------------------------------------------------
enum { EPI_F16 = 0, EPI_GELU_F16 = 1, EPI_F32 = 2, EPI_KT_F16 = 3 };

template <int EPI>
__global__ __launch_bounds__(256) void gemm_f16_kernel(
    const _Float16* __restrict__ A, int lda,
    const _Float16* __restrict__ W,
    const float* __restrict__ bias,
    void* __restrict__ Cout, int ldc,
    int Mdim, int Ndim, int Kdim) {
  constexpr int BM = 128, BN = 128, BK = 32;
  constexpr int LAS = BK + 8;   // 40 halves (80B rows, 16B aligned)
  constexpr int LBS = BN + 8;   // 136 halves (272B rows, 16B aligned)
  __shared__ _Float16 sA[2][BM * LAS];   // 2 x 10 KB
  __shared__ _Float16 sB[2][BK * LBS];   // 2 x 8.5 KB

  const int tid  = threadIdx.x;
  const int lane = tid & 31;
  const int wid  = tid >> 5;
  const int wm   = wid & 3;
  const int wn   = wid >> 2;
  const int m0   = blockIdx.y * BM;
  const int n0   = blockIdx.x * BN;

  v8f acc[2][4] = {};

  const int ar = tid >> 1;        // 0..127
  const int ap = (tid & 1) << 4;  // 0 / 16 halves
  const int br = tid >> 3;        // 0..31
  const int bp = (tid & 7) << 4;  // 0..112 halves

  auto stage = [&](int bufi, int k0) {
    const _Float16* asrc = A + (size_t)(m0 + ar) * lda + k0 + ap;
    async_b128(&sA[bufi][ar * LAS + ap],     asrc);
    async_b128(&sA[bufi][ar * LAS + ap + 8], asrc + 8);
    const _Float16* bsrc = W + (size_t)(k0 + br) * Ndim + n0 + bp;
    async_b128(&sB[bufi][br * LBS + bp],     bsrc);
    async_b128(&sB[bufi][br * LBS + bp + 8], bsrc + 8);
  };

  stage(0, 0);
  wait_async0();
  __syncthreads();

  int buf = 0;
  for (int k0 = 0; k0 < Kdim; k0 += BK) {
    const int nbuf = buf ^ 1;
    if (k0 + BK < Kdim) stage(nbuf, k0 + BK);  // async prefetch next tile

    v16h af[2], bf[4];
#pragma unroll
    for (int i = 0; i < 2; ++i)
      af[i] = load_a_frag(&sA[buf][(wm * 32 + i * 16) * LAS], LAS, lane);
#pragma unroll
    for (int j = 0; j < 4; ++j)
      bf[j] = load_b_frag(&sB[buf][wn * 64 + j * 16], LBS, lane);
#pragma unroll
    for (int i = 0; i < 2; ++i)
#pragma unroll
      for (int j = 0; j < 4; ++j)
        acc[i][j] = wmma_f16(af[i], bf[j], acc[i][j]);

    wait_async0();     // this wave's DMA into nbuf done
    __syncthreads();   // everyone done reading buf + all DMAs visible
    buf = nbuf;
  }

  // C/D layout: lane holds column n = lane&15; VGPR r holds row r + 8*(lane>>4)
  const int nloc  = lane & 15;
  const int mbase = (lane >> 4) << 3;
#pragma unroll
  for (int i = 0; i < 2; ++i) {
#pragma unroll
    for (int j = 0; j < 4; ++j) {
      const int col = n0 + wn * 64 + j * 16 + nloc;
      const float bv = bias ? bias[col] : 0.0f;
#pragma unroll
      for (int r = 0; r < 8; ++r) {
        const int row = m0 + wm * 32 + i * 16 + mbase + r;
        float val = acc[i][j][r] + bv;
        if (EPI == EPI_GELU_F16) val = gelu_exact(val);
        if (EPI == EPI_F32) {
          ((float*)Cout)[(size_t)row * ldc + col] = val;
        } else if (EPI == EPI_KT_F16) {
          // Scatter as kT[(b*NH+h)*HD+d][m] so attention's q*k^T WMMA can
          // read B fragments as contiguous rows (no gathers / TR loads).
          const int bb = row >> 9, mm = row & 511;
          const int hh = col >> 6, dd = col & 63;
          ((_Float16*)Cout)[(size_t)((bb * NH_ + hh) * HD_ + dd) * M_ + mm] =
              (_Float16)val;
        } else {
          ((_Float16*)Cout)[(size_t)row * ldc + col] = (_Float16)val;
        }
      }
    }
  }
}

// ---------------------------------------------------------------------------
// Fused attention: per block = (head h, 16 s-rows, batch b).
// Pass 1: logits = q*kT/8 via WMMA into LDS (f32).
// Softmax (masked) in f32; probs written back as f16 (aliasing logits LDS);
// mem_attn accumulated with global f32 atomics (mean over heads).
// Pass 2: attended = probs @ v via WMMA with a 2-way K-split (4 d-tiles x 2 K-halves).
// Chunk staging uses async global->LDS DMA.
// ---------------------------------------------------------------------------
__global__ __launch_bounds__(256) void attn_kernel(
    const _Float16* __restrict__ q,    // ROWS x H
    const _Float16* __restrict__ kT,   // (B*NH*HD) x M
    const _Float16* __restrict__ v,    // MROWS x H
    const int* __restrict__ mask,      // B x M
    _Float16* __restrict__ combined,   // ROWS x 2H (writes cols H..2H-1)
    float* __restrict__ mem_attn) {    // B x S x M
  __shared__ float    logits[16 * 512];  // 32KB; aliased: probs f16 + partials
  __shared__ _Float16 qt[16 * 64];       // 2KB
  __shared__ _Float16 stage[9216];       // 18KB (kT chunk 64x128 / v 128x64)
  __shared__ float    red[16][17];

  _Float16* probs = (_Float16*)logits;   // floats [0..4095] as 16x512 f16
  float*    attp  = logits + 4096;       // floats [4096..6143]: 2x16x64 partials

  const int h   = blockIdx.x;
  const int s0  = blockIdx.y * 16;
  const int b   = blockIdx.z;
  const int tid = threadIdx.x;
  const int lane = tid & 31;
  const int wid  = tid >> 5;

  // load q tile 16x64 (plain 8-byte load + LDS store per thread)
  {
    const int r  = tid >> 4;
    const int c0 = (tid & 15) << 2;
    const _Float16* src = q + (size_t)(b * S_ + s0 + r) * H_ + h * HD_ + c0;
    // 4 halves per thread: cheapest as one b64-equivalent; use two rounds of
    // b128 would over-fetch, so keep a plain 8-byte copy here.
    *(uint64_t*)(qt + r * 64 + c0) = *(const uint64_t*)src;
  }

  // ---- pass 1: logits ----
  for (int mc = 0; mc < 4; ++mc) {
    const int m0 = mc * 128;
    __syncthreads();
    {  // stage kT chunk: 64(d) x 128(m), stride 136 halves
      const int d = tid >> 2;
      const int p = (tid & 3) << 5;
      const _Float16* src = kT + (size_t)((b * NH_ + h) * HD_ + d) * M_ + m0 + p;
      _Float16* dst = stage + d * 136 + p;
      async_b128(dst,      src);
      async_b128(dst + 8,  src + 8);
      async_b128(dst + 16, src + 16);
      async_b128(dst + 24, src + 24);
    }
    wait_async0();
    __syncthreads();
    v8f acc = {};
#pragma unroll
    for (int kk = 0; kk < 64; kk += 32) {
      v16h a  = load_a_frag(qt + kk, 64, lane);
      v16h bf = load_b_frag(stage + kk * 136 + wid * 16, 136, lane);
      acc = wmma_f16(a, bf, acc);
    }
    const int nloc  = lane & 15;
    const int mbase = (lane >> 4) << 3;
#pragma unroll
    for (int r = 0; r < 8; ++r)
      logits[(mbase + r) * 512 + m0 + wid * 16 + nloc] = acc[r];
  }
  __syncthreads();

  // ---- softmax (16 threads per row, 32 cols each) ----
  {
    const int r  = tid >> 4;
    const int li = tid & 15;
    const int c0 = li * 32;
    float e[32];
    float mx = -3.402823e38f;
#pragma unroll
    for (int c = 0; c < 32; ++c) {
      const int col = c0 + c;
      const float x = (mask[b * M_ + col] != 0)
                          ? logits[r * 512 + col] * 0.125f   // 1/sqrt(HD)
                          : -3.402823e38f;
      e[c] = x;
      mx = fmaxf(mx, x);
    }
    red[r][li] = mx;
    __syncthreads();
    if (li == 0) {
      float m2 = red[r][0];
      for (int i = 1; i < 16; ++i) m2 = fmaxf(m2, red[r][i]);
      red[r][16] = m2;
    }
    __syncthreads();
    mx = red[r][16];
    float sum = 0.f;
#pragma unroll
    for (int c = 0; c < 32; ++c) { e[c] = __expf(e[c] - mx); sum += e[c]; }
    __syncthreads();
    red[r][li] = sum;
    __syncthreads();
    if (li == 0) {
      float s2 = 0.f;
      for (int i = 0; i < 16; ++i) s2 += red[r][i];
      red[r][16] = s2;
    }
    __syncthreads();
    const float inv = 1.0f / red[r][16];
    float* matt = mem_attn + (size_t)(b * S_ + s0 + r) * M_ + c0;
#pragma unroll
    for (int c = 0; c < 32; ++c) {
      const float p = e[c] * inv;  // surprise renorm == identity, elided
      e[c] = p;
      atomicAdd(matt + c, p * (1.0f / NH_));
    }
    __syncthreads();  // all f32 logits reads done; safe to alias as f16 probs
#pragma unroll
    for (int c = 0; c < 32; ++c)
      probs[r * 512 + c0 + c] = (_Float16)e[c];
  }
  __syncthreads();

  // ---- pass 2: attended = probs @ v  (wave = (d-tile jt, K-half h2)) ----
  const int jt = wid & 3;
  const int h2 = wid >> 2;
  v8f acc2 = {};
  for (int mc = 0; mc < 4; ++mc) {
    const int m0 = mc * 128;
    __syncthreads();
    {  // stage v chunk: 128(m) x 64(d), stride 72 halves
      const int ml = tid >> 1;
      const int p  = (tid & 1) << 5;
      const _Float16* src = v + (size_t)(b * M_ + m0 + ml) * H_ + h * HD_ + p;
      _Float16* dst = stage + ml * 72 + p;
      async_b128(dst,      src);
      async_b128(dst + 8,  src + 8);
      async_b128(dst + 16, src + 16);
      async_b128(dst + 24, src + 24);
    }
    wait_async0();
    __syncthreads();
#pragma unroll
    for (int kk = 0; kk < 64; kk += 32) {
      v16h a  = load_a_frag(probs + m0 + h2 * 64 + kk, 512, lane);
      v16h bf = load_b_frag(stage + (h2 * 64 + kk) * 72 + jt * 16, 72, lane);
      acc2 = wmma_f16(a, bf, acc2);
    }
  }
  {
    const int nloc  = lane & 15;
    const int mbase = (lane >> 4) << 3;
#pragma unroll
    for (int r = 0; r < 8; ++r)
      attp[(h2 * 16 + mbase + r) * 64 + jt * 16 + nloc] = acc2[r];
  }
  __syncthreads();
  {  // reduce the two K-halves and write into combined[:, H + h*64 + d]
    const int idx0 = tid * 4;
#pragma unroll
    for (int u = 0; u < 4; ++u) {
      const int idx = idx0 + u;
      const int rr  = idx >> 6;
      const int dd  = idx & 63;
      const float val = attp[idx] + attp[1024 + idx];
      combined[(size_t)(b * S_ + s0 + rr) * (2 * H_) + H_ + h * HD_ + dd] =
          (_Float16)val;
    }
  }
}

// ---------------------------------------------------------------------------
// Gate: sigmoid(g1 @ Wg2 + bg2), N==1 -> per-row wave32 reduction.
// ---------------------------------------------------------------------------
__global__ __launch_bounds__(256) void gate_kernel(
    const _Float16* __restrict__ g1, const float* __restrict__ Wg2,
    const float* __restrict__ bg2, float* __restrict__ gate) {
  const int lane = threadIdx.x & 31;
  const int wid  = threadIdx.x >> 5;
  const int row  = blockIdx.x * 8 + wid;
  float s = 0.f;
  for (int c = lane; c < H_; c += 32)
    s += (float)g1[(size_t)row * H_ + c] * Wg2[c];
#pragma unroll
  for (int off = 16; off > 0; off >>= 1) s += __shfl_down(s, off, 32);
  if (lane == 0) gate[row] = 1.0f / (1.0f + __expf(-(s + bg2[0])));
}

// ---------------------------------------------------------------------------
// Final: out = gate * LayerNorm(fpre) + (1-gate) * hidden_states  (f32)
// ---------------------------------------------------------------------------
__global__ __launch_bounds__(256) void final_kernel(
    const float* __restrict__ fpre, const float* __restrict__ hs,
    const float* __restrict__ gate, const float* __restrict__ lng,
    const float* __restrict__ lnb, float* __restrict__ out) {
  __shared__ float rsum[8], rsq[8], sh[2];
  const int row = blockIdx.x;
  const int tid = threadIdx.x;
  const int lane = tid & 31, wid = tid >> 5;
  const float* x = fpre + (size_t)row * H_;
  float s = 0.f, q = 0.f, vals[4];
#pragma unroll
  for (int u = 0; u < 4; ++u) {
    const float xv = x[tid + u * 256];
    vals[u] = xv; s += xv; q += xv * xv;
  }
#pragma unroll
  for (int off = 16; off > 0; off >>= 1) {
    s += __shfl_down(s, off, 32);
    q += __shfl_down(q, off, 32);
  }
  if (lane == 0) { rsum[wid] = s; rsq[wid] = q; }
  __syncthreads();
  if (tid == 0) {
    float ts = 0.f, tq = 0.f;
    for (int i = 0; i < 8; ++i) { ts += rsum[i]; tq += rsq[i]; }
    const float mu = ts / H_;
    sh[0] = mu;
    sh[1] = rsqrtf(tq / H_ - mu * mu + 1e-5f);
  }
  __syncthreads();
  const float mu = sh[0], rstd = sh[1];
  const float g = gate[row];
#pragma unroll
  for (int u = 0; u < 4; ++u) {
    const int c = tid + u * 256;
    const float y = (vals[u] - mu) * rstd * lng[c] + lnb[c];
    out[(size_t)row * H_ + c] = g * y + (1.0f - g) * hs[(size_t)row * H_ + c];
  }
}

// ---------------------------------------------------------------------------
extern "C" void kernel_launch(void* const* d_in, const int* in_sizes, int n_in,
                              void* d_out, int out_size, void* d_ws,
                              size_t ws_size, hipStream_t stream) {
  (void)in_sizes; (void)n_in; (void)out_size; (void)ws_size;
  const float* hs   = (const float*)d_in[0];
  const float* mem  = (const float*)d_in[1];
  const int*   mask = (const int*)d_in[2];
  // d_in[3] surprise_score: exact no-op in the reference (softmax renorm).
  const float* Wq = (const float*)d_in[4];  const float* bq = (const float*)d_in[5];
  const float* Wk = (const float*)d_in[6];  const float* bk = (const float*)d_in[7];
  const float* Wv = (const float*)d_in[8];  const float* bv = (const float*)d_in[9];
  const float* W1 = (const float*)d_in[10]; const float* b1 = (const float*)d_in[11];
  const float* W2 = (const float*)d_in[12]; const float* b2 = (const float*)d_in[13];
  const float* lng = (const float*)d_in[14]; const float* lnb = (const float*)d_in[15];
  const float* Wg1 = (const float*)d_in[16]; const float* bg1 = (const float*)d_in[17];
  const float* Wg2 = (const float*)d_in[18]; const float* bg2 = (const float*)d_in[19];

  char* ws = (char*)d_ws;
  size_t off = 0;
  auto take = [&](size_t bytes) -> char* {
    char* p = ws + off;
    off += (bytes + 255) & ~(size_t)255;
    return p;
  };
  _Float16* combined = (_Float16*)take((size_t)ROWS * 2 * H_ * 2);
  _Float16* Wq16  = (_Float16*)take((size_t)H_ * H_ * 2);
  _Float16* Wk16  = (_Float16*)take((size_t)HQ_ * H_ * 2);
  _Float16* Wv16  = (_Float16*)take((size_t)HQ_ * H_ * 2);
  _Float16* W116  = (_Float16*)take((size_t)2 * H_ * H_ * 2);
  _Float16* W216  = (_Float16*)take((size_t)H_ * H_ * 2);
  _Float16* Wg116 = (_Float16*)take((size_t)2 * H_ * H_ * 2);
  _Float16* mem16 = (_Float16*)take((size_t)MROWS * HQ_ * 2);
  _Float16* q16   = (_Float16*)take((size_t)ROWS * H_ * 2);
  _Float16* kT16  = (_Float16*)take((size_t)B_ * NH_ * HD_ * M_ * 2);
  _Float16* v16   = (_Float16*)take((size_t)MROWS * H_ * 2);
  _Float16* h116  = (_Float16*)take((size_t)ROWS * H_ * 2);
  _Float16* g116  = (_Float16*)take((size_t)ROWS * H_ * 2);
  float*    fpre  = (float*)take((size_t)ROWS * H_ * 4);
  float*    gatep = (float*)take((size_t)ROWS * 4);

  auto cdiv = [](int a, int b) { return (a + b - 1) / b; };
  const dim3 blk(256);

  // casts (hidden_states lands in combined[:, 0:H] at stride 2H -> free concat)
  cast_kernel<<<cdiv(ROWS * H_, 256), blk, 0, stream>>>(hs, combined, ROWS * H_, H_, 2 * H_);
  cast_kernel<<<cdiv(H_ * H_, 256), blk, 0, stream>>>(Wq, Wq16, H_ * H_, H_ * H_, H_ * H_);
  cast_kernel<<<cdiv(HQ_ * H_, 256), blk, 0, stream>>>(Wk, Wk16, HQ_ * H_, HQ_ * H_, HQ_ * H_);
  cast_kernel<<<cdiv(HQ_ * H_, 256), blk, 0, stream>>>(Wv, Wv16, HQ_ * H_, HQ_ * H_, HQ_ * H_);
  cast_kernel<<<cdiv(2 * H_ * H_, 256), blk, 0, stream>>>(W1, W116, 2 * H_ * H_, 2 * H_ * H_, 2 * H_ * H_);
  cast_kernel<<<cdiv(H_ * H_, 256), blk, 0, stream>>>(W2, W216, H_ * H_, H_ * H_, H_ * H_);
  cast_kernel<<<cdiv(2 * H_ * H_, 256), blk, 0, stream>>>(Wg1, Wg116, 2 * H_ * H_, 2 * H_ * H_, 2 * H_ * H_);
  cast_kernel<<<cdiv(MROWS * HQ_, 256), blk, 0, stream>>>(mem, mem16, MROWS * HQ_, MROWS * HQ_, MROWS * HQ_);

  // projections
  gemm_f16_kernel<EPI_F16><<<dim3(H_ / 128, ROWS / 128), blk, 0, stream>>>(
      combined, 2 * H_, Wq16, bq, q16, H_, ROWS, H_, H_);
  gemm_f16_kernel<EPI_KT_F16><<<dim3(H_ / 128, MROWS / 128), blk, 0, stream>>>(
      mem16, HQ_, Wk16, bk, kT16, 0, MROWS, H_, HQ_);
  gemm_f16_kernel<EPI_F16><<<dim3(H_ / 128, MROWS / 128), blk, 0, stream>>>(
      mem16, HQ_, Wv16, bv, v16, H_, MROWS, H_, HQ_);

  // attention (+ mem_attn output, zeroed first: atomics accumulate into d_out)
  float* mem_attn = (float*)d_out + (size_t)ROWS * H_;
  zero_kernel<<<cdiv(ROWS * M_, 256), blk, 0, stream>>>(mem_attn, ROWS * M_);
  attn_kernel<<<dim3(NH_, S_ / 16, B_), blk, 0, stream>>>(q16, kT16, v16, mask,
                                                          combined, mem_attn);

  // gate branch
  gemm_f16_kernel<EPI_GELU_F16><<<dim3(H_ / 128, ROWS / 128), blk, 0, stream>>>(
      combined, 2 * H_, Wg116, bg1, g116, H_, ROWS, H_, 2 * H_);
  gate_kernel<<<ROWS / 8, blk, 0, stream>>>(g116, Wg2, bg2, gatep);

  // fused branch
  gemm_f16_kernel<EPI_GELU_F16><<<dim3(H_ / 128, ROWS / 128), blk, 0, stream>>>(
      combined, 2 * H_, W116, b1, h116, H_, ROWS, H_, 2 * H_);
  gemm_f16_kernel<EPI_F32><<<dim3(H_ / 128, ROWS / 128), blk, 0, stream>>>(
      h116, H_, W216, b2, fpre, H_, ROWS, H_, H_);

  // LayerNorm + gated mix (reads fp32 hidden_states for the residual path)
  final_kernel<<<ROWS, blk, 0, stream>>>(fpre, hs, gatep, lng, lnb, (float*)d_out);
}
